// MyModel_40681930227839
// MI455X (gfx1250) — matrix operand order repaired and checked by reading, not compiled, not run on hardware
//
#include <hip/hip_runtime.h>

// ---------------- model constants ----------------
constexpr int Bsz = 32, Ssz = 256, Dm = 768, Hh = 12, DHd = 64, FFd = 3072;
constexpr int Ll = 12, SENSEn = 64, ATTn = 64;
constexpr int Ntok = Bsz * Ssz;
constexpr int DD  = Dm * Dm;    // 589824
constexpr int DFF = Dm * FFd;   // 2359296
#define LN_EPS 1e-12f

typedef _Float16 v16h __attribute__((ext_vector_type(16)));
typedef _Float16 v8h  __attribute__((ext_vector_type(8)));
typedef float    v8f  __attribute__((ext_vector_type(8)));
typedef int      v4i  __attribute__((vector_size(16)));   // matches builtin proto

#define AS1 __attribute__((address_space(1)))
#define AS3 __attribute__((address_space(3)))

// gfx1250 async global->LDS path (guarded; falls back to ds_store_b128)
#if defined(__HIP_DEVICE_COMPILE__) && \
    __has_builtin(__builtin_amdgcn_global_load_async_to_lds_b128)
#define HAVE_ASYNC_LDS 1
#else
#define HAVE_ASYNC_LDS 0
#endif

#if HAVE_ASYNC_LDS
#if __has_builtin(__builtin_amdgcn_s_wait_asynccnt)
#define WAIT_ASYNC() __builtin_amdgcn_s_wait_asynccnt(0)
#else
#define WAIT_ASYNC() asm volatile("s_wait_asynccnt 0" ::: "memory")
#endif
#endif

// -------- f32 [K][N] -> f16 [N][K] transpose+convert (32x32 LDS tiles) -------
__global__ __launch_bounds__(256) void convt_kernel(
    const float* __restrict__ src, _Float16* __restrict__ dst, int K, int N)
{
    __shared__ _Float16 tile[32][33];
    int kt = blockIdx.y * 32, nt = blockIdx.x * 32;
    int tx = threadIdx.x;   // 0..31
    int ty = threadIdx.y;   // 0..7
#pragma unroll
    for (int i = 0; i < 32; i += 8)
        tile[ty + i][tx] = (_Float16)src[(size_t)(kt + ty + i) * N + nt + tx];
    __syncthreads();
#pragma unroll
    for (int i = 0; i < 32; i += 8)
        dst[(size_t)(nt + ty + i) * K + kt + tx] = tile[tx][ty + i];
}

// ---------------- WMMA GEMM: C[M,N] = A[M,K](f32) @ W(f16, pre-transposed
// [N][K]) (+bias)(+gelu). Block tile 128x64, K-step 32. 256 threads = 8 waves
// (4x2); each wave owns a 32x32 patch = 2x2 v_wmma_f32_16x16x32_f16 tiles.
// A: register-pipelined f32->f16 staging. B: async global->LDS double-buffered
// copy (ASYNCcnt) when available, else vector ds_store_b128.
#define TM 128
#define TN 64
#define TK 32
#define LDA 40   // padded LDS row stride (f16 elems); 80B = 16B-aligned rows
#define LDB 40

__global__ __launch_bounds__(256) void gemm_f16_wmma(
    const float* __restrict__ A, const _Float16* __restrict__ WT,
    const float* __restrict__ bias, float* __restrict__ C,
    int M, int N, int K, int act)
{
    __shared__ _Float16 lA[TM * LDA];        // [m][k]   (10 KB)
    __shared__ _Float16 lB[2][TN * LDB];     // [n][k] double-buffered (2x5 KB)

    const int t    = threadIdx.x;
    const int lane = t & 31;
    const int wv   = t >> 5;
    const int wm   = wv & 3;      // wave row slab (32 rows)
    const int wn   = wv >> 2;     // wave col slab (32 cols)
    const int l16  = lane & 15;
    const int hl   = lane >> 4;   // lane half (ISA 16-bit operand layout)
    const int m0   = blockIdx.y * TM;
    const int n0   = blockIdx.x * TN;

    // A staging: 2 threads per row, 16 consecutive floats each (4x b128 loads)
    const int ar  = t >> 1;            // 0..127
    const int ac  = (t & 1) * 16;      // 0 or 16
    const int agr = m0 + ar;
    const bool azero = (agr >= M);
    const int agrc = azero ? (M - 1) : agr;   // clamp: loads always in-bounds
    const float4* Abase = (const float4*)(A + (size_t)agrc * K + ac);

    // B staging: WT is [N][K]; 4 threads per n-row, 8 consecutive halves each
    const int bn  = t >> 2;            // 0..63
    const int bkc = (t & 3) * 8;       // 0,8,16,24
    const _Float16* Wrow = WT + (size_t)(n0 + bn) * K + bkc;

    float4 ra0, ra1, ra2, ra3;
#if !HAVE_ASYNC_LDS
    v8h rb;
#endif

#define LOAD_A(KK)                                                            \
    do {                                                                      \
        const float4* ap_ = Abase + ((KK) >> 2);                              \
        ra0 = ap_[0]; ra1 = ap_[1]; ra2 = ap_[2]; ra3 = ap_[3];               \
    } while (0)

#define STORE_A()                                                             \
    do {                                                                      \
        float4 q0 = ra0, q1 = ra1, q2 = ra2, q3 = ra3;                        \
        if (azero) {                                                          \
            q0 = make_float4(0.f, 0.f, 0.f, 0.f);                             \
            q1 = q0; q2 = q0; q3 = q0;                                        \
        }                                                                     \
        v8h h0, h1;                                                           \
        h0[0] = (_Float16)q0.x; h0[1] = (_Float16)q0.y;                       \
        h0[2] = (_Float16)q0.z; h0[3] = (_Float16)q0.w;                       \
        h0[4] = (_Float16)q1.x; h0[5] = (_Float16)q1.y;                       \
        h0[6] = (_Float16)q1.z; h0[7] = (_Float16)q1.w;                       \
        h1[0] = (_Float16)q2.x; h1[1] = (_Float16)q2.y;                       \
        h1[2] = (_Float16)q2.z; h1[3] = (_Float16)q2.w;                       \
        h1[4] = (_Float16)q3.x; h1[5] = (_Float16)q3.y;                       \
        h1[6] = (_Float16)q3.z; h1[7] = (_Float16)q3.w;                       \
        *(v8h*)&lA[ar * LDA + ac]     = h0;                                   \
        *(v8h*)&lA[ar * LDA + ac + 8] = h1;                                   \
    } while (0)

#if HAVE_ASYNC_LDS
#define ISSUE_B(KK, PB)                                                       \
    __builtin_amdgcn_global_load_async_to_lds_b128(                           \
        (AS1 v4i*)(Wrow + (KK)),                                              \
        (AS3 v4i*)(&lB[(PB)][bn * LDB + bkc]),                                \
        0, 0)
#endif

    v8f acc[2][2];
#pragma unroll
    for (int i = 0; i < 2; ++i)
#pragma unroll
        for (int j = 0; j < 2; ++j)
#pragma unroll
            for (int e = 0; e < 8; ++e) acc[i][j][e] = 0.0f;

    LOAD_A(0);
#if HAVE_ASYNC_LDS
    ISSUE_B(0, 0);
#else
    rb = *(const v8h*)(Wrow + 0);
#endif

    for (int kk = 0; kk < K; kk += TK) {
        const int pb = (kk / TK) & 1;
        STORE_A();
#if HAVE_ASYNC_LDS
        WAIT_ASYNC();                      // our wave's B copy has landed
#else
        *(v8h*)&lB[pb][bn * LDB + bkc] = rb;
#endif
        __syncthreads();                   // all waves' lA + lB[pb] visible
        if (kk + TK < K) {
            LOAD_A(kk + TK);               // overlap next loads with WMMAs
#if HAVE_ASYNC_LDS
            ISSUE_B(kk + TK, pb ^ 1);      // async into the other buffer
#else
            rb = *(const v8h*)(Wrow + kk + TK);
#endif
        }
        if (kk + 2 * TK < K)               // gfx1250 prefetch hint
            __builtin_prefetch(Wrow + kk + 2 * TK, 0, 0);

        // A 16x32 fragment: lane<16 -> K {0..15}; lane>=16 -> {16..31}
        v16h af[2], bfr[2];
#pragma unroll
        for (int tm = 0; tm < 2; ++tm) {
            const _Float16* ap = &lA[(wm * 32 + tm * 16 + l16) * LDA];
#pragma unroll
            for (int i = 0; i < 16; ++i) {
                af[tm][i]     = ap[hl * 16 + i];
            }
        }
        // B fragment (n-major in LDS): lane half covers K = hl*16 .. hl*16+15
#pragma unroll
        for (int tn = 0; tn < 2; ++tn) {
            const _Float16* bp = &lB[pb][(wn * 32 + tn * 16 + l16) * LDB];
#pragma unroll
            for (int i = 0; i < 16; ++i) bfr[tn][i] = bp[hl * 16 + i];
        }

#pragma unroll
        for (int tm = 0; tm < 2; ++tm)
#pragma unroll
            for (int tn = 0; tn < 2; ++tn)
                acc[tm][tn] = __builtin_amdgcn_wmma_f32_16x16x32_f16(
                    false, af[tm], false, bfr[tn],
                    (short)0, acc[tm][tn], false, false);
        __syncthreads();                   // lA reuse / lB[pb^1] still private
    }

    // C/D layout: VGPR r -> lanes 0-15 M=r, lanes 16-31 M=r+8
#pragma unroll
    for (int tm = 0; tm < 2; ++tm)
#pragma unroll
        for (int tn = 0; tn < 2; ++tn)
#pragma unroll
            for (int r = 0; r < 8; ++r) {
                int row = m0 + wm * 32 + tm * 16 + r + hl * 8;
                int col = n0 + wn * 32 + tn * 16 + l16;
                if (row < M) {
                    float v = acc[tm][tn][r];
                    if (bias) v += bias[col];
                    if (act == 1)  // exact GELU
                        v = 0.5f * v * (1.0f + erff(v * 0.70710678118654752f));
                    C[(size_t)row * N + col] = v;
                }
            }
#undef LOAD_A
#undef STORE_A
#ifdef ISSUE_B
#undef ISSUE_B
#endif
}

// ---------------- embeddings + LayerNorm (one block per token) ----------------
__global__ __launch_bounds__(256) void embed_ln_kernel(
    const int* __restrict__ ids, const int* __restrict__ tts,
    const float* __restrict__ we, const float* __restrict__ pe,
    const float* __restrict__ te, const float* __restrict__ g,
    const float* __restrict__ bb, float* __restrict__ x)
{
    __shared__ float buf[Dm];
    __shared__ float red[256];
    int tok = blockIdx.x;
    int s   = tok % Ssz;
    int id  = ids[tok];
    int tt  = tts[tok];
    int t   = threadIdx.x;

    float ps = 0.0f;
    for (int i = t; i < Dm; i += 256) {
        float v = we[(size_t)id * Dm + i] + pe[(size_t)s * Dm + i] +
                  te[(size_t)tt * Dm + i];
        buf[i] = v;
        ps += v;
    }
    red[t] = ps; __syncthreads();
    for (int o = 128; o > 0; o >>= 1) { if (t < o) red[t] += red[t + o]; __syncthreads(); }
    float mu = red[0] / (float)Dm;
    __syncthreads();
    float pv = 0.0f;
    for (int i = t; i < Dm; i += 256) { float d = buf[i] - mu; pv += d * d; }
    red[t] = pv; __syncthreads();
    for (int o = 128; o > 0; o >>= 1) { if (t < o) red[t] += red[t + o]; __syncthreads(); }
    float rstd = rsqrtf(red[0] / (float)Dm + LN_EPS);
    for (int i = t; i < Dm; i += 256)
        x[(size_t)tok * Dm + i] = (buf[i] - mu) * rstd * g[i] + bb[i];
}

// ---------------- residual add + LayerNorm (in place on x) ----------------
__global__ __launch_bounds__(256) void add_ln_kernel(
    float* __restrict__ x, const float* __restrict__ tadd,
    const float* __restrict__ g, const float* __restrict__ bb)
{
    __shared__ float buf[Dm];
    __shared__ float red[256];
    int tok = blockIdx.x;
    int t   = threadIdx.x;

    float ps = 0.0f;
    for (int i = t; i < Dm; i += 256) {
        float v = x[(size_t)tok * Dm + i] + tadd[(size_t)tok * Dm + i];
        buf[i] = v;
        ps += v;
    }
    red[t] = ps; __syncthreads();
    for (int o = 128; o > 0; o >>= 1) { if (t < o) red[t] += red[t + o]; __syncthreads(); }
    float mu = red[0] / (float)Dm;
    __syncthreads();
    float pv = 0.0f;
    for (int i = t; i < Dm; i += 256) { float d = buf[i] - mu; pv += d * d; }
    red[t] = pv; __syncthreads();
    for (int o = 128; o > 0; o >>= 1) { if (t < o) red[t] += red[t + o]; __syncthreads(); }
    float rstd = rsqrtf(red[0] / (float)Dm + LN_EPS);
    for (int i = t; i < Dm; i += 256)
        x[(size_t)tok * Dm + i] = (buf[i] - mu) * rstd * g[i] + bb[i];
}

// ---------------- attention: one block per (b, h, q-row); 256 threads = keys --
__global__ __launch_bounds__(256) void attn_kernel(
    const float* __restrict__ q, const float* __restrict__ k,
    const float* __restrict__ v, const int* __restrict__ mask,
    float* __restrict__ out)
{
    __shared__ float qrow[DHd];
    __shared__ float p[Ssz];
    __shared__ float red[256];
    int qi = blockIdx.x, h = blockIdx.y, b = blockIdx.z;
    int t  = threadIdx.x;

    if (t < DHd) qrow[t] = q[((size_t)(b * Ssz + qi)) * Dm + h * DHd + t];
    __syncthreads();

    const float4* kp4 =
        (const float4*)(k + ((size_t)(b * Ssz + t)) * Dm + h * DHd);
    float s = 0.0f;
#pragma unroll
    for (int i = 0; i < DHd / 4; ++i) {
        float4 kv = kp4[i];
        s += qrow[4 * i + 0] * kv.x + qrow[4 * i + 1] * kv.y +
             qrow[4 * i + 2] * kv.z + qrow[4 * i + 3] * kv.w;
    }
    s *= 0.125f;                                     // 1/sqrt(64)
    s += (1.0f - (float)mask[b * Ssz + t]) * -10000.0f;

    red[t] = s; __syncthreads();
    for (int o = 128; o > 0; o >>= 1) { if (t < o) red[t] = fmaxf(red[t], red[t + o]); __syncthreads(); }
    float m = red[0]; __syncthreads();
    float e = __expf(s - m);
    p[t] = e; red[t] = e; __syncthreads();
    for (int o = 128; o > 0; o >>= 1) { if (t < o) red[t] += red[t + o]; __syncthreads(); }
    float z = red[0];

    if (t < DHd) {
        float acc = 0.0f;
        for (int j = 0; j < Ssz; ++j)
            acc += p[j] * v[((size_t)(b * Ssz + j)) * Dm + h * DHd + t];
        out[((size_t)(b * Ssz + qi)) * Dm + h * DHd + t] = acc / z;
    }
}

// ---------------- pun gather + projection through att_Wq ----------------
__global__ __launch_bounds__(64) void pq_kernel(
    const float* __restrict__ x, const int* __restrict__ loc,
    const float* __restrict__ wq, float* __restrict__ pq)
{
    int b = blockIdx.x, n = threadIdx.x;
    int row = b * Ssz + loc[b];
    const float* xr = x + (size_t)row * Dm;
    float acc = 0.0f;
    for (int d = 0; d < Dm; ++d) acc += xr[d] * wq[(size_t)d * ATTn + n];
    pq[b * ATTn + n] = acc;
}

// ---------------- sense scores + top-2 indices ----------------
__global__ __launch_bounds__(64) void top2_kernel(
    const float* __restrict__ pq, const float* __restrict__ sk,
    float* __restrict__ out)
{
    __shared__ float sc[SENSEn];
    int b = blockIdx.x, j = threadIdx.x;
    float acc = 0.0f;
    for (int d = 0; d < ATTn; ++d) acc += pq[b * ATTn + d] * sk[j * ATTn + d];
    sc[j] = acc * 0.125f;                            // /sqrt(64); softmax monotone
    __syncthreads();
    if (j == 0) {
        int i0 = 0; float v0 = sc[0];
        for (int i = 1; i < SENSEn; ++i) if (sc[i] > v0) { v0 = sc[i]; i0 = i; }
        int i1 = (i0 == 0) ? 1 : 0; float v1 = sc[i1];
        for (int i = 0; i < SENSEn; ++i)
            if (i != i0 && sc[i] > v1) { v1 = sc[i]; i1 = i; }
        out[b * 2 + 0] = (float)i0;
        out[b * 2 + 1] = (float)i1;
    }
}

// ---------------- host side ----------------
static inline void launch_gemm(const float* A, const _Float16* WT,
                               const float* bias, float* C,
                               int M, int N, int K, int act, hipStream_t stream)
{
    dim3 grid(N / TN, (M + TM - 1) / TM);
    gemm_f16_wmma<<<grid, dim3(256), 0, stream>>>(A, WT, bias, C, M, N, K, act);
}

static inline void launch_convt(const float* src, _Float16* dst, int K, int N,
                                hipStream_t stream)
{
    dim3 grid(N / 32, K / 32);
    convt_kernel<<<grid, dim3(32, 8), 0, stream>>>(src, dst, K, N);
}

extern "C" void kernel_launch(void* const* d_in, const int* in_sizes, int n_in,
                              void* d_out, int out_size, void* d_ws, size_t ws_size,
                              hipStream_t stream)
{
    const int*   input_ids  = (const int*)  d_in[0];
    const int*   token_type = (const int*)  d_in[1];
    const int*   attn_mask  = (const int*)  d_in[2];
    const int*   location   = (const int*)  d_in[3];
    const float* sense_emb  = (const float*)d_in[4];
    const float* word_emb   = (const float*)d_in[5];
    const float* pos_emb    = (const float*)d_in[6];
    const float* type_emb   = (const float*)d_in[7];
    const float* emb_g      = (const float*)d_in[8];
    const float* emb_b      = (const float*)d_in[9];
    const float* Wq  = (const float*)d_in[10];
    const float* bq  = (const float*)d_in[11];
    const float* Wk  = (const float*)d_in[12];
    const float* bk  = (const float*)d_in[13];
    const float* Wv  = (const float*)d_in[14];
    const float* bv  = (const float*)d_in[15];
    const float* Wo  = (const float*)d_in[16];
    const float* bo  = (const float*)d_in[17];
    const float* ln1g = (const float*)d_in[18];
    const float* ln1b = (const float*)d_in[19];
    const float* W1  = (const float*)d_in[20];
    const float* b1  = (const float*)d_in[21];
    const float* W2  = (const float*)d_in[22];
    const float* b2  = (const float*)d_in[23];
    const float* ln2g = (const float*)d_in[24];
    const float* ln2b = (const float*)d_in[25];
    const float* attWq = (const float*)d_in[26];
    const float* attWk = (const float*)d_in[27];

    // workspace: X | T0 | T1 | R(QKV / FFN-mid) | SK | PQ | WH(f16 weights, T)
    const size_t ND  = (size_t)Ntok * Dm;     //  6,291,456 floats
    const size_t NFF = (size_t)Ntok * FFd;    // 25,165,824 floats
    float* X  = (float*)d_ws;
    float* T0 = X  + ND;
    float* T1 = T0 + ND;
    float* R  = T1 + ND;                      // 3*ND <= NFF; region size NFF
    float* SK = R  + NFF;
    float* PQ = SK + (size_t)SENSEn * ATTn;
    _Float16* WH = (_Float16*)(PQ + (size_t)Bsz * ATTn);
    float* Q  = R;
    float* Kb = R + ND;
    float* Vb = R + 2 * ND;

    _Float16* wq16 = WH;                       // each stored transposed [N][K]
    _Float16* wk16 = WH + (size_t)DD;
    _Float16* wv16 = WH + (size_t)2 * DD;
    _Float16* wo16 = WH + (size_t)3 * DD;
    _Float16* w116 = WH + (size_t)4 * DD;
    _Float16* w216 = WH + (size_t)4 * DD + DFF;

    (void)in_sizes; (void)n_in; (void)out_size; (void)ws_size;

    // embeddings + LN
    embed_ln_kernel<<<Ntok, 256, 0, stream>>>(input_ids, token_type, word_emb,
                                              pos_emb, type_emb, emb_g, emb_b, X);

    for (int l = 0; l < Ll; ++l) {
        // stage this layer's weights as transposed f16 once
        launch_convt(Wq + (size_t)l * DD,  wq16, Dm,  Dm,  stream);
        launch_convt(Wk + (size_t)l * DD,  wk16, Dm,  Dm,  stream);
        launch_convt(Wv + (size_t)l * DD,  wv16, Dm,  Dm,  stream);
        launch_convt(Wo + (size_t)l * DD,  wo16, Dm,  Dm,  stream);
        launch_convt(W1 + (size_t)l * DFF, w116, Dm,  FFd, stream);
        launch_convt(W2 + (size_t)l * DFF, w216, FFd, Dm,  stream);

        launch_gemm(X, wq16, bq + (size_t)l * Dm, Q,  Ntok, Dm, Dm, 0, stream);
        launch_gemm(X, wk16, bk + (size_t)l * Dm, Kb, Ntok, Dm, Dm, 0, stream);
        launch_gemm(X, wv16, bv + (size_t)l * Dm, Vb, Ntok, Dm, Dm, 0, stream);

        attn_kernel<<<dim3(Ssz, Hh, Bsz), 256, 0, stream>>>(Q, Kb, Vb,
                                                            attn_mask, T0);

        launch_gemm(T0, wo16, bo + (size_t)l * Dm, T1, Ntok, Dm, Dm, 0, stream);
        add_ln_kernel<<<Ntok, 256, 0, stream>>>(X, T1, ln1g + (size_t)l * Dm,
                                                ln1b + (size_t)l * Dm);

        launch_gemm(X, w116, b1 + (size_t)l * FFd, R,  Ntok, FFd, Dm, 1, stream);
        launch_gemm(R, w216, b2 + (size_t)l * Dm,  T1, Ntok, Dm, FFd, 0, stream);
        add_ln_kernel<<<Ntok, 256, 0, stream>>>(X, T1, ln2g + (size_t)l * Dm,
                                                ln2b + (size_t)l * Dm);
    }

    // sense head: SK = sense_emb @ att_Wk  (64x64, K=768)
    launch_convt(attWk, wq16, Dm, ATTn, stream);      // reuse WH staging
    launch_gemm(sense_emb, wq16, nullptr, SK, SENSEn, ATTn, Dm, 0, stream);
    pq_kernel<<<Bsz, 64, 0, stream>>>(X, location, attWq, PQ);
    top2_kernel<<<Bsz, 64, 0, stream>>>(PQ, SK, (float*)d_out);
}
